// MoEDecoder_29257317220859
// MI455X (gfx1250) — matrix operand order repaired and checked by reading, not compiled, not run on hardware
//
#include <hip/hip_runtime.h>

// MoE decoder, CDNA5 (gfx1250), wave32.
//   - all GEMMs via v_wmma_f32_16x16x32_bf16 (f32 accumulate)
//   - f32 weights stay in global (36MB << 192MB L2), converted to bf16 while
//     staging tiles into LDS; next-chunk rows prefetched (global_prefetch_b8)
//   - activations in bf16 ping-pong workspace buffers
//   - routed + shared expert GEMMs fused per 64-token tile (route constant
//     within a 256-token batch element), chained WMMA accumulation
//   - 64x64 block tile, 8 waves, 2 accumulators/wave -> A fragment reused
//     across 2-4 WMMAs per LDS read
//   - compile-time template flags -> branch-free inner loops

typedef __bf16 bf16;
typedef __bf16 v16bf __attribute__((ext_vector_type(16)));
typedef __bf16 v8bf  __attribute__((ext_vector_type(8)));
typedef float  v8f   __attribute__((ext_vector_type(8)));

#define BM 64
#define BN 64
#define BK 64
#define LDP (BK + 8)   // padded LDS row: 72 elems = 144B = 9*16B (16B-aligned frags)

union ABfrag { v16bf v; v8bf h[2]; };

// stage a 64x64 f32 tile -> bf16 LDS tile (row-major [row][k]); 16 elems/thread
__device__ __forceinline__ void stage_f32(bf16 (*dst)[LDP], const float* __restrict__ src_base,
                                          int rBase, int K, int k0, int srow, int scol) {
    const float* src = &src_base[(size_t)(rBase + srow) * K + k0 + scol];
    bf16* d = &dst[srow][scol];
#pragma unroll
    for (int c = 0; c < 16; c += 4) {
        float4 p = *(const float4*)(src + c);
        d[c + 0] = (bf16)p.x; d[c + 1] = (bf16)p.y;
        d[c + 2] = (bf16)p.z; d[c + 3] = (bf16)p.w;
    }
}

template<bool AF32, bool HAS_W1, bool RES, bool RELU, bool OUTBF>
__global__ __launch_bounds__(256)
void moe_gemm(const float* __restrict__ Af32,   // f32 activations (proj) or null
              const bf16*  __restrict__ Abf,    // bf16 activations or null
              int K,
              const float* __restrict__ W0,     // [Nout, K] f32 (shared / only weight)
              const float* __restrict__ b0,     // [Nout] f32
              const float* __restrict__ W1,     // routed base [E, Nout, K] f32
              const float* __restrict__ b1,     // routed bias base [E, Nout] f32
              const int*   __restrict__ route,  // per-batch expert id
              const bf16*  __restrict__ Rbf,    // residual [N, Nout] bf16
              bf16* __restrict__ Obf, float* __restrict__ Of32,
              int Nout) {
    __shared__ bf16 As [BM][LDP];
    __shared__ bf16 Bs0[BN][LDP];
    __shared__ bf16 Bs1[HAS_W1 ? BN : 1][LDP];

    const int rowBase = blockIdx.y * BM;        // token tile base
    const int nBase   = blockIdx.x * BN;        // out-feature tile base

    const int lane = threadIdx.x & 31;
    const int wave = threadIdx.x >> 5;          // 0..7
    const int wm   = wave & 3;                  // M quarter (0..3)
    const int wn   = wave >> 2;                 // N half (0..1)
    const int l15  = lane & 15;
    const int lhi  = lane >> 4;                 // 0/1

    const int srow = threadIdx.x >> 2;          // staging row 0..63
    const int scol = (threadIdx.x & 3) * 16;    // staging col 0,16,32,48

    // routed expert for this token tile (route constant within 256-token batch)
    const float* W1e = nullptr;
    const float* b1e = nullptr;
    if constexpr (HAS_W1) {
        const int e = route[rowBase >> 8];      // 256 tokens per batch element
        W1e = W1 + (size_t)e * Nout * K;
        b1e = b1 + (size_t)e * Nout;
    }

    const int nG0 = nBase + wn * 32 + l15;      // first output column of this wave
    const int nG1 = nG0 + 16;                   // second output column
    float bias0 = b0[nG0], bias1 = b0[nG1];
    if constexpr (HAS_W1) { bias0 += b1e[nG0]; bias1 += b1e[nG1]; }
    v8f acc0, acc1;
#pragma unroll
    for (int i = 0; i < 8; ++i) { acc0[i] = bias0; acc1[i] = bias1; }

    const int aRow  = wm * 16 + l15;
    const int bRow0 = wn * 32 + l15;
    const int bRow1 = bRow0 + 16;

    for (int k0 = 0; k0 < K; k0 += BK) {
        // prefetch next K-chunk weight rows into cache while this chunk computes
        if (k0 + BK < K) {
            __builtin_prefetch(&W0[(size_t)(nBase + srow) * K + k0 + BK + scol], 0, 0);
            if constexpr (HAS_W1)
                __builtin_prefetch(&W1e[(size_t)(nBase + srow) * K + k0 + BK + scol], 0, 0);
        }
        __syncthreads();
        // ---- stage A tile (64 x 64) ----
        if constexpr (AF32) {
            stage_f32(As, Af32, rowBase, K, k0, srow, scol);
        } else {
            const bf16* src = &Abf[(size_t)(rowBase + srow) * K + k0 + scol];
            *(v8bf*)&As[srow][scol]     = *(const v8bf*)(src);
            *(v8bf*)&As[srow][scol + 8] = *(const v8bf*)(src + 8);
        }
        // ---- stage weight tiles (64 x 64 each), f32 -> bf16 ----
        stage_f32(Bs0, W0, nBase, K, k0, srow, scol);
        if constexpr (HAS_W1) stage_f32(Bs1, W1e, nBase, K, k0, srow, scol);
        __syncthreads();

        // ---- WMMA over the K chunk; A fragment reused across 2-4 WMMAs ----
#pragma unroll
        for (int kk = 0; kk < BK; kk += 32) {
            ABfrag a, b;
            // A 16x32 bf16 layout: lane = row, 16 contiguous K per half-wave
            a.h[0] = *(const v8bf*)&As[aRow][kk + lhi * 16];
            a.h[1] = *(const v8bf*)&As[aRow][kk + lhi * 16 + 8];
            // B 32x16 bf16 layout: lane = column, 16 contiguous K per half-wave
            b.h[0] = *(const v8bf*)&Bs0[bRow0][kk + lhi * 16];
            b.h[1] = *(const v8bf*)&Bs0[bRow0][kk + lhi * 16 + 8];
            acc0 = __builtin_amdgcn_wmma_f32_16x16x32_bf16(
                false, a.v, false, b.v, (short)0, acc0, false, false);
            b.h[0] = *(const v8bf*)&Bs0[bRow1][kk + lhi * 16];
            b.h[1] = *(const v8bf*)&Bs0[bRow1][kk + lhi * 16 + 8];
            acc1 = __builtin_amdgcn_wmma_f32_16x16x32_bf16(
                false, a.v, false, b.v, (short)0, acc1, false, false);
            if constexpr (HAS_W1) {
                b.h[0] = *(const v8bf*)&Bs1[bRow0][kk + lhi * 16];
                b.h[1] = *(const v8bf*)&Bs1[bRow0][kk + lhi * 16 + 8];
                acc0 = __builtin_amdgcn_wmma_f32_16x16x32_bf16(
                    false, a.v, false, b.v, (short)0, acc0, false, false);
                b.h[0] = *(const v8bf*)&Bs1[bRow1][kk + lhi * 16];
                b.h[1] = *(const v8bf*)&Bs1[bRow1][kk + lhi * 16 + 8];
                acc1 = __builtin_amdgcn_wmma_f32_16x16x32_bf16(
                    false, a.v, false, b.v, (short)0, acc1, false, false);
            }
        }
    }

    // ---- epilogue: residual + ReLU + store (C layout: VGPR i -> rows i / i+8) ----
#pragma unroll
    for (int i = 0; i < 8; ++i) {
        const int mG = rowBase + wm * 16 + lhi * 8 + i;
        float v0 = acc0[i], v1 = acc1[i];
        if constexpr (RES) {
            v0 += (float)Rbf[(size_t)mG * Nout + nG0];
            v1 += (float)Rbf[(size_t)mG * Nout + nG1];
        }
        if constexpr (RELU) { v0 = fmaxf(v0, 0.0f); v1 = fmaxf(v1, 0.0f); }
        if constexpr (OUTBF) {
            Obf[(size_t)mG * Nout + nG0] = (bf16)v0;
            Obf[(size_t)mG * Nout + nG1] = (bf16)v1;
        } else {
            Of32[(size_t)mG * Nout + nG0] = v0;
            Of32[(size_t)mG * Nout + nG1] = v1;
        }
    }
}

extern "C" void kernel_launch(void* const* d_in, const int* in_sizes, int n_in,
                              void* d_out, int out_size, void* d_ws, size_t ws_size,
                              hipStream_t stream) {
    (void)in_sizes; (void)n_in; (void)out_size; (void)ws_size;
    const float* x     = (const float*)d_in[0];
    const int*   route = (const int*)  d_in[1];
    const float* Wi    = (const float*)d_in[2];
    const float* bi    = (const float*)d_in[3];
    const float* Wr    = (const float*)d_in[4];   // [2,16,512,512]
    const float* br    = (const float*)d_in[5];   // [2,16,512]
    const float* Ws    = (const float*)d_in[6];   // [2,1,512,512]
    const float* bs    = (const float*)d_in[7];   // [2,1,512]
    const float* Wo    = (const float*)d_in[8];
    const float* bo    = (const float*)d_in[9];
    float* out = (float*)d_out;

    const int N   = 64 * 256;   // 16384 tokens
    const int HID = 512;

    bf16* hA = (bf16*)d_ws;                       // [N, 512] bf16
    bf16* hB = hA + (size_t)N * HID;              // [N, 512] bf16

    dim3 block(256);
    dim3 gridH(HID / BN, N / BM);                 // (8, 256)
    dim3 gridO(256 / BN, N / BM);                 // (4, 256)

    // projector: hA = relu(x @ Wi^T + bi)   K=256
    moe_gemm<true, false, false, true, true><<<gridH, block, 0, stream>>>(
        x, nullptr, 256, Wi, bi, nullptr, nullptr, nullptr, nullptr,
        hA, nullptr, HID);
    // layer 0: hB = relu(hA@Ws0^T + bs0 + hA@Wr0[e]^T + br0[e] + hA)   K=512
    moe_gemm<false, true, true, true, true><<<gridH, block, 0, stream>>>(
        nullptr, hA, HID,
        Ws + (size_t)0 * HID * HID, bs + 0 * HID,
        Wr + (size_t)0 * 16 * HID * HID, br + 0 * 16 * HID,
        route, hA, hB, nullptr, HID);
    // layer 1: hA = relu(hB@Ws1^T + bs1 + hB@Wr1[e]^T + br1[e] + hB)
    moe_gemm<false, true, true, true, true><<<gridH, block, 0, stream>>>(
        nullptr, hB, HID,
        Ws + (size_t)1 * HID * HID, bs + 1 * HID,
        Wr + (size_t)1 * 16 * HID * HID, br + 1 * 16 * HID,
        route, hB, hA, nullptr, HID);
    // output: out = hA @ Wo^T + bo  (f32, no relu)
    moe_gemm<false, false, false, false, false><<<gridO, block, 0, stream>>>(
        nullptr, hA, HID, Wo, bo, nullptr, nullptr, nullptr, nullptr,
        nullptr, out, 256);
}
